// RNN_4363686773193
// MI455X (gfx1250) — matrix operand order repaired and checked by reading, not compiled, not run on hardware
//
#include <hip/hip_runtime.h>

// ---------------------------------------------------------------------------
// RNN: xp = x@Wx + b  (big parallel GEMM, bf16 WMMA)
//      h_t = sigmoid(xp_t + h_{t-1}@Wh)  (sequential scan, 1 WGP, WMMA matvec)
//      out = h_T@Wo + bo
// T=65536, F=512, H=256, O=128.  xp (64 MB f32) lives in d_ws (L2-resident).
// ---------------------------------------------------------------------------

typedef __bf16  bf16_t;
typedef bf16_t  v16bf __attribute__((ext_vector_type(16)));
typedef bf16_t  v2bf  __attribute__((ext_vector_type(2)));
typedef float   v8f   __attribute__((ext_vector_type(8)));
typedef unsigned v8u  __attribute__((ext_vector_type(8)));

// Pack two f32 -> packed bf16 dword.  Prefer the HW packed convert if the
// builtin is declared; otherwise use native __bf16 casts and let the
// backend pick the best instruction sequence.
#if __has_builtin(__builtin_amdgcn_cvt_pk_bf16_f32)
__device__ __forceinline__ unsigned pack2_bf16(float a, float b) {
    auto r = __builtin_amdgcn_cvt_pk_bf16_f32(a, b);
    return __builtin_bit_cast(unsigned, r);
}
#else
__device__ __forceinline__ unsigned pack2_bf16(float a, float b) {
    v2bf v;
    v[0] = (bf16_t)a;
    v[1] = (bf16_t)b;
    return __builtin_bit_cast(unsigned, v);
}
#endif
__device__ __forceinline__ unsigned short bf16_1(float a) {
    bf16_t h = (bf16_t)a;
    return __builtin_bit_cast(unsigned short, h);
}

__device__ __forceinline__ v8f wmma_bf16(v8u a, v8u b, v8f c) {
    return __builtin_amdgcn_wmma_f32_16x16x32_bf16(
        false, __builtin_bit_cast(v16bf, a),
        false, __builtin_bit_cast(v16bf, b),
        (short)0, c, false, false);
}

// --- CDNA5 async copy global->LDS (ISA §15.18, GLOBAL_LOAD_ASYNC_TO_LDS_B32,
// GV mode: per-lane 64-bit global addr, per-lane LDS byte address in VDST).
// Verified to assemble on this toolchain in the previous round.
__device__ __forceinline__ void async_g2l_b32(const float* lds_ptr, const float* gptr) {
    unsigned lds = (unsigned)(uintptr_t)lds_ptr;   // low 32 bits = LDS offset
    asm volatile("global_load_async_to_lds_b32 %0, %1, off"
                 :: "v"(lds), "v"(gptr) : "memory");
}
__device__ __forceinline__ void wait_asynccnt_le6() {
#if __has_builtin(__builtin_amdgcn_s_wait_asynccnt)
    __builtin_amdgcn_s_wait_asynccnt(6);
#else
    asm volatile("s_wait_asynccnt 0x6" ::: "memory");
#endif
    asm volatile("" ::: "memory");   // keep LDS reads below the wait
}

// ---------------------------------------------------------------------------
// Phase 1: xp[t][n] = sum_k x[t][k]*W[k][n] + b[n]   (t<65536, k<512, n<256)
// One wave computes a 64x16 output strip (4 M-subtiles sharing each B tile),
// K in 16 chunks of 32 (bf16 WMMA).
// grid = (1024, 2), block = 256 (8 waves); wave w -> N-tile blockIdx.y*8+w.
// ---------------------------------------------------------------------------
__global__ void __launch_bounds__(256) gemm_xp_kernel(
    const float* __restrict__ x, const float* __restrict__ W,
    const float* __restrict__ b, float* __restrict__ xp)
{
    const int lane = threadIdx.x & 31;
    const int wave = threadIdx.x >> 5;
    const int l15  = lane & 15;
    const int hi   = lane >> 4;
    const int mt   = blockIdx.x;              // M strip (0..1023) = rows 64*mt..
    const int nt   = blockIdx.y * 8 + wave;   // N tile (0..15)
    const int col  = nt * 16 + l15;           // B/D column this lane supplies

    v8f acc[4];
    #pragma unroll
    for (int mm = 0; mm < 4; ++mm) acc[mm] = (v8f){0.f,0.f,0.f,0.f,0.f,0.f,0.f,0.f};

    #pragma unroll
    for (int c = 0; c < 16; ++c) {
        // B 32x16 bf16 layout: lane l15 = col N; lanes 0-15 K=0..15,
        // lanes 16-31 K=16..31; VGPR v: K = 2v (+16hi) and +1.
        v8u Bv;
        #pragma unroll
        for (int v = 0; v < 8; ++v) {
            const int k0 = 32 * c + 16 * hi + 2 * v;
            Bv[v] = pack2_bf16(W[(size_t)k0 * 256 + col],
                               W[(size_t)(k0 + 1) * 256 + col]);
        }
        // A 16x32 bf16 layout: lane l15 = row M; VGPR v: K=2v+16*hi (+1).
        // => floats [32c+16hi .. +16).
        #pragma unroll
        for (int mm = 0; mm < 4; ++mm) {
            const int arow = mt * 64 + mm * 16 + l15;
            const float* xrow = x + (size_t)arow * 512;
            const float4* pa = (const float4*)(xrow + 32 * c + 16 * hi);
            float4 a0 = pa[0], a1 = pa[1];                 // K = 0..7   (+16hi)
            const float4* pb = (const float4*)(xrow + 32 * c + 8 + 16 * hi);
            float4 a2 = pb[0], a3 = pb[1];                 // K = 8..15  (+16hi)
            v8u A;
            A[0] = pack2_bf16(a0.x, a0.y); A[1] = pack2_bf16(a0.z, a0.w);
            A[2] = pack2_bf16(a1.x, a1.y); A[3] = pack2_bf16(a1.z, a1.w);
            A[4] = pack2_bf16(a2.x, a2.y); A[5] = pack2_bf16(a2.z, a2.w);
            A[6] = pack2_bf16(a3.x, a3.y); A[7] = pack2_bf16(a3.z, a3.w);
            acc[mm] = wmma_bf16(A, Bv, acc[mm]);
        }
    }

    const float bias = b[col];
    // D layout: VGPR r holds M = r (lanes 0-15) / M = 8+r (lanes 16-31).
    #pragma unroll
    for (int mm = 0; mm < 4; ++mm) {
        #pragma unroll
        for (int r = 0; r < 8; ++r) {
            const int orow = mt * 64 + mm * 16 + hi * 8 + r;
            xp[(size_t)orow * 256 + col] = acc[mm][r] + bias;
        }
    }
}

// ---------------------------------------------------------------------------
// Phase 2+3: sequential scan on a single workgroup (8 waves, one WGP).
// Wave w owns output columns [32w,32w+32) as two 16-col WMMA N-tiles; its
// Wh columns are pinned in 128 VGPRs/lane (bf16, B layout).  h is exchanged
// through a double-buffered bf16 LDS array.  xp rows stream through an LDS
// ring filled by ASYNCcnt-tracked global_load_async_to_lds_b32 with a
// 6-step lead (ring is wave-private: wave w's lanes fetch exactly the
// columns wave w consumes).  Four independent WMMA accumulator chains
// (2 N-tiles x 2 K-halves) keep the per-step dependency chain short.
// ---------------------------------------------------------------------------
__global__ void __launch_bounds__(256, 1) rnn_scan_kernel(
    const float* __restrict__ xp, const float* __restrict__ W,
    const float* __restrict__ Wo, const float* __restrict__ bo,
    float* __restrict__ out)
{
    __shared__ __align__(16) unsigned hb[2][128];  // bf16 h, double buffered
    __shared__ float ring[8][256];                 // xp row ring buffer
    __shared__ float hf[256];                      // final h in f32

    const int tid  = threadIdx.x;
    const int lane = tid & 31;
    const int wave = tid >> 5;
    const int l15  = lane & 15;
    const int hi   = lane >> 4;
    const int T    = 65536;
    const float* Wh = W + 512 * 256;               // Wh = W[F:, :]

    // --- Pin Wh in registers: bf16, WMMA B layout, 2 N-tiles x 8 K-chunks.
    v8u Bw[2][8];
    #pragma unroll
    for (int ntile = 0; ntile < 2; ++ntile) {
        const int col = 32 * wave + 16 * ntile + l15;
        #pragma unroll
        for (int c = 0; c < 8; ++c) {
            v8u t;
            #pragma unroll
            for (int v = 0; v < 8; ++v) {
                const int k0 = 32 * c + 16 * hi + 2 * v;
                t[v] = pack2_bf16(Wh[(size_t)k0 * 256 + col],
                                  Wh[(size_t)(k0 + 1) * 256 + col]);
            }
            Bw[ntile][c] = t;
        }
    }

    // --- h0 = 0; prime the async xp ring with rows 0..5 (slots 0..5).
    if (tid < 128) hb[0][tid] = 0u;
    #pragma unroll
    for (int r = 0; r < 6; ++r)
        async_g2l_b32(&ring[r][tid], xp + (size_t)r * 256 + tid);
    __syncthreads();

    for (int t = 0; t < T; ++t) {
        // Issue async fill of row t+6 (clamped); 1 op/step, in-order retire.
        int lrow = t + 6; if (lrow > T - 1) lrow = T - 1;
        async_g2l_b32(&ring[(t + 6) & 7][tid], xp + (size_t)lrow * 256 + tid);
        int frow = t + 64; if (frow > T - 1) frow = T - 1;
        __builtin_prefetch(xp + (size_t)frow * 256 + tid, 0, 0);

        // Matvec: A = h broadcast into all 16 M-rows; 4 independent
        // accumulator chains (2 N-tiles x 2 K-halves) hide WMMA latency.
        const unsigned* hsrc = hb[t & 1];
        v8f acc0a = {0.f,0.f,0.f,0.f,0.f,0.f,0.f,0.f};
        v8f acc0b = acc0a, acc1a = acc0a, acc1b = acc0a;
        #pragma unroll
        for (int c = 0; c < 4; ++c) {
            const uint4 lo0 = *(const uint4*)(hsrc + 16 * c + 8 * hi);
            const uint4 hi0 = *(const uint4*)(hsrc + 16 * c + 4 + 8 * hi);
            const uint4 lo1 = *(const uint4*)(hsrc + 16 * (c + 4) + 8 * hi);
            const uint4 hi1 = *(const uint4*)(hsrc + 16 * (c + 4) + 4 + 8 * hi);
            v8u Aa, Ab;
            Aa[0] = lo0.x; Aa[1] = lo0.y; Aa[2] = lo0.z; Aa[3] = lo0.w;
            Aa[4] = hi0.x; Aa[5] = hi0.y; Aa[6] = hi0.z; Aa[7] = hi0.w;
            Ab[0] = lo1.x; Ab[1] = lo1.y; Ab[2] = lo1.z; Ab[3] = lo1.w;
            Ab[4] = hi1.x; Ab[5] = hi1.y; Ab[6] = hi1.z; Ab[7] = hi1.w;
            acc0a = wmma_bf16(Aa, Bw[0][c], acc0a);
            acc1a = wmma_bf16(Aa, Bw[1][c], acc1a);
            acc0b = wmma_bf16(Ab, Bw[0][c + 4], acc0b);
            acc1b = wmma_bf16(Ab, Bw[1][c + 4], acc1b);
        }

        // Row t of xp landed >= 6 steps ago: ASYNCcnt <= 6 guarantees it.
        wait_asynccnt_le6();

        // Row M=0 of D lives in VGPR0 of lanes 0..15.  Apply sigmoid, publish.
        if (lane < 16) {
            const int c0 = 32 * wave + l15;
            const int c1 = c0 + 16;
            const float z0 = ring[t & 7][c0] + acc0a[0] + acc0b[0];
            const float z1 = ring[t & 7][c1] + acc1a[0] + acc1b[0];
            const float h0 = 1.f / (1.f + __expf(-z0));
            const float h1 = 1.f / (1.f + __expf(-z1));
            unsigned short* hd = (unsigned short*)hb[(t + 1) & 1];
            hd[c0] = bf16_1(h0);
            hd[c1] = bf16_1(h1);
            if (t == T - 1) { hf[c0] = h0; hf[c1] = h1; }
        }
        __syncthreads();
    }

    // --- out = hT @ Wo + bo  (tiny: 128 threads x 256 MACs)
    if (tid < 128) {
        float acc = bo[tid];
        #pragma unroll 8
        for (int j = 0; j < 256; ++j) acc += hf[j] * Wo[(size_t)j * 128 + tid];
        out[tid] = acc;
    }
}

// ---------------------------------------------------------------------------
extern "C" void kernel_launch(void* const* d_in, const int* in_sizes, int n_in,
                              void* d_out, int out_size, void* d_ws, size_t ws_size,
                              hipStream_t stream) {
    const float* x  = (const float*)d_in[0];   // 65536 x 512
    const float* W  = (const float*)d_in[1];   // 768 x 256
    const float* b  = (const float*)d_in[2];   // 256
    const float* Wo = (const float*)d_in[3];   // 256 x 128
    const float* bo = (const float*)d_in[4];   // 128
    float* out = (float*)d_out;                // 128
    float* xp  = (float*)d_ws;                 // 65536 x 256 f32 = 64 MB scratch

    dim3 g1(1024, 2);
    gemm_xp_kernel<<<g1, 256, 0, stream>>>(x, W, b, xp);
    rnn_scan_kernel<<<1, 256, 0, stream>>>(xp, W, Wo, bo, out);
}
